// RnnCell_30348238913895
// MI455X (gfx1250) — matrix operand order repaired and checked by reading, not compiled, not run on hardware
//
#include <hip/hip_runtime.h>
#include <hip/hip_bf16.h>

// Problem constants (from reference): B=32, T=2048, D=512, RED=2
#define BATCH 32
#define TP    1024            // T / RED
#define INW   1024            // D * RED  (LSTM input width)
#define HD    512             // hidden
#define G4    2048            // 4*H gate width
#define PD    512             // projection
#define MROWS (BATCH * TP)    // 32768 rows for the big GEMMs

typedef __attribute__((ext_vector_type(16))) __bf16          v16bf;
typedef __attribute__((ext_vector_type(2)))  __bf16          v2bf;
typedef __attribute__((ext_vector_type(16))) unsigned short  v16us;
typedef __attribute__((ext_vector_type(8)))  unsigned short  v8us;
typedef __attribute__((ext_vector_type(8)))  float           v8f;
typedef __attribute__((ext_vector_type(4)))  float           f32x4;
typedef __attribute__((ext_vector_type(4)))  unsigned int    u32x4;

#if defined(__has_builtin)
#if __has_builtin(__builtin_amdgcn_cvt_pk_bf16_f32)
#define HAVE_PK_BF16 1
#endif
#if __has_builtin(__builtin_amdgcn_tanhf)
#define HAVE_HW_TANH 1
#endif
#endif

__device__ __forceinline__ unsigned short f2bf(float f) {
  unsigned u = __builtin_bit_cast(unsigned, f);
  u += 0x7FFFu + ((u >> 16) & 1u);           // round-to-nearest-even
  return (unsigned short)(u >> 16);
}
__device__ __forceinline__ float bf2f(unsigned short h) {
  unsigned u = ((unsigned)h) << 16;
  return __builtin_bit_cast(float, u);
}
// Packed f32x2 -> bf16x2: one V_CVT_PK_BF16_F32 when available.
__device__ __forceinline__ v2bf pk2bf(float a, float b) {
#if HAVE_PK_BF16
  auto pr = __builtin_amdgcn_cvt_pk_bf16_f32(a, b);
  return __builtin_bit_cast(v2bf, pr);
#else
  unsigned u = (unsigned)f2bf(a) | ((unsigned)f2bf(b) << 16);
  return __builtin_bit_cast(v2bf, u);
#endif
}
__device__ __forceinline__ float hw_tanh(float x) {
#if HAVE_HW_TANH
  return __builtin_amdgcn_tanhf(x);          // V_TANH_F32
#else
  return tanhf(x);
#endif
}
__device__ __forceinline__ float sigm(float x) {
  return 0.5f * hw_tanh(0.5f * x) + 0.5f;    // sigmoid via hardware tanh
}

__device__ __forceinline__ v8f wmma_bf16(v16bf a, v16bf b, v8f c) {
  return __builtin_amdgcn_wmma_f32_16x16x32_bf16(false, a, false, b, (short)0, c,
                                                 false, false);
}

// ---------------------------------------------------------------------------
// Pack a row-major f32 weight W[n][k] (n = output col of W^T GEMM, k = reduce)
// into the CDNA5 WMMA B-operand register layout as bf16 tiles:
//   dst[((nt*ktiles + kt)*32 + lane)*16 + e] = W[nt*16 + lane%16][kt*32 + 8*(lane/16) + e + (e >= 8 ? 8 : 0)]
// ---------------------------------------------------------------------------
__global__ void pack_b_kernel(const float* __restrict__ src,
                              unsigned short* __restrict__ dst,
                              int ntiles, int ktiles, int Kcols) {
  int idx = blockIdx.x * blockDim.x + threadIdx.x;
  int total = ntiles * ktiles * 32;
  if (idx >= total) return;
  int L  = idx & 31;
  int kt = (idx >> 5) % ktiles;
  int nt = idx / (ktiles * 32);
  int n  = nt * 16 + (L & 15);
  int kb = kt * 32 + (L >> 4) * 8;
  const float* s = src + (size_t)n * Kcols + kb;
  union { v16bf v; v2bf p[8]; } u;
#pragma unroll
  for (int e = 0; e < 8; ++e) u.p[e] = pk2bf(s[2 * e + (e >= 4 ? 8 : 0)], s[2 * e + 1 + (e >= 4 ? 8 : 0)]);
  *(v16bf*)(dst + (size_t)idx * 16) = u.v;
}

// ---------------------------------------------------------------------------
// init: zero out_cat, reset barrier counters, emit lens to the output tail
// ---------------------------------------------------------------------------
__global__ void init_kernel(unsigned int* __restrict__ outcat_u32, size_t n_u32,
                            unsigned int* __restrict__ syncc,
                            const int* __restrict__ input_size,
                            float* __restrict__ lens_out) {
  size_t i = (size_t)blockIdx.x * blockDim.x + threadIdx.x;
  size_t stride = (size_t)gridDim.x * blockDim.x;
  for (size_t k = i; k < n_u32; k += stride) outcat_u32[k] = 0u;
  if (i < 4) syncc[i] = 0u;
  if (i < BATCH) lens_out[i] = (float)(input_size[i] / 2);
}

// ---------------------------------------------------------------------------
// xg = x @ w_ih^T  (bf16 in / f32 accumulate / bf16 out). x is the contiguous
// reinterpretation of input_data as (MROWS x INW). One launch per direction.
// Each wave: one 16(M)x64(N) strip, K=1024 in 32 k-tiles of 32.
// ---------------------------------------------------------------------------
__global__ void xgemm_kernel(const float* __restrict__ x,
                             const unsigned short* __restrict__ wp,  // packed B
                             unsigned short* __restrict__ xg) {
  const int lane = threadIdx.x & 31;
  const int wave = threadIdx.x >> 5;
  const int mt     = blockIdx.x >> 2;
  const int nchunk = blockIdx.x & 3;
  const int half = lane >> 4;
  const int mrow = mt * 16 + (lane & 15);
  const float* xrow = x + (size_t)mrow * INW;

  v8f zf = {0.f,0.f,0.f,0.f,0.f,0.f,0.f,0.f};
  v8f acc[4] = {zf, zf, zf, zf};
  const int ntb = nchunk * 32 + wave * 4;

  for (int kt = 0; kt < 32; ++kt) {
    int kb = kt * 32 + 8 * half;               // A runs: [kb,kb+8) and [kb+16,kb+24)
    __builtin_prefetch(xrow + kb + 64, 0, 1);  // global_prefetch for next tiles
    f32x4 r0 = *(const f32x4*)(xrow + kb);
    f32x4 r1 = *(const f32x4*)(xrow + kb + 4);
    f32x4 r2 = *(const f32x4*)(xrow + kb + 16);
    f32x4 r3 = *(const f32x4*)(xrow + kb + 20);
    union { v16bf v; v2bf p[8]; } ua;
    ua.p[0] = pk2bf(r0[0], r0[1]);
    ua.p[1] = pk2bf(r0[2], r0[3]);
    ua.p[2] = pk2bf(r1[0], r1[1]);
    ua.p[3] = pk2bf(r1[2], r1[3]);
    ua.p[4] = pk2bf(r2[0], r2[1]);
    ua.p[5] = pk2bf(r2[2], r2[3]);
    ua.p[6] = pk2bf(r3[0], r3[1]);
    ua.p[7] = pk2bf(r3[2], r3[3]);
    v16bf a = ua.v;
#pragma unroll
    for (int j = 0; j < 4; ++j) {
      int nt = ntb + j;
      v16bf b = *(const v16bf*)(wp + ((size_t)(nt * 32 + kt) * 32 + lane) * 16);
      acc[j] = wmma_bf16(a, b, acc[j]);
    }
  }
#pragma unroll
  for (int j = 0; j < 4; ++j) {
    int n = (ntb + j) * 16 + (lane & 15);
#pragma unroll
    for (int r = 0; r < 8; ++r) {
      int m = mt * 16 + r + 8 * half;
      xg[(size_t)m * G4 + n] = f2bf(acc[j][r]);
    }
  }
}

// ---------------------------------------------------------------------------
// Persistent bidirectional LSTM recurrence.
// 32 blocks x 256 threads: blocks [0,16) forward, [16,32) reverse.
// Block `slice` owns h columns [32*slice, 32*slice+32) and the 4 matching
// gate chunks. Its 128KB packed w_hh slice lives in LDS for all 1024 steps.
// Cross-WGP h exchange: double-buffered bf16 h in global (L2-resident) +
// release/acquire atomic counter barrier each step.
// ---------------------------------------------------------------------------
#define REC_SMEM 188544
__global__ void lstm_rec_kernel(const unsigned short* __restrict__ xg_all,
                                const unsigned short* __restrict__ whh_p_all,
                                const float* __restrict__ b_ih_f,
                                const float* __restrict__ b_hh_f,
                                const float* __restrict__ b_ih_r,
                                const float* __restrict__ b_hh_r,
                                const int* __restrict__ input_size,
                                unsigned short* __restrict__ out_cat,
                                unsigned short* __restrict__ hglob,
                                unsigned int* __restrict__ syncc) {
  extern __shared__ char smem[];
  unsigned short* Bs = (unsigned short*)(smem);            // 8 nt x 16 kt x 32 x 16 bf16 = 128KB
  unsigned short* As = (unsigned short*)(smem + 131072);   // 2 mt x 16 kt x 32 x 16 bf16 = 32KB
  float* gbuf = (float*)(smem + 163840);                   // [4 gate][32 col][32 b] = 16KB
  float* cbuf = (float*)(smem + 180224);                   // [32 col][32 b]
  float* hbuf = (float*)(smem + 184320);                   // [32 col][32 b]
  int*   lensS = (int*)(smem + 188416);                    // [32]

  const int tid   = threadIdx.x;
  const int lane  = tid & 31;
  const int wave  = tid >> 5;
  const int dir   = blockIdx.x >> 4;
  const int slice = blockIdx.x & 15;

  const unsigned short* xg    = xg_all + (size_t)dir * MROWS * G4;
  const unsigned short* whh_p = whh_p_all + (size_t)dir * (128 * 16 * 512);
  const float* bih = dir ? b_ih_r : b_ih_f;
  const float* bhh = dir ? b_hh_r : b_hh_f;

  if (tid < BATCH) lensS[tid] = input_size[tid] / 2;

  // Stage this block's w_hh slice (8 N-tiles x 16 K-tiles) into LDS.
  for (int i = 0; i < 32; ++i) {
    int o = i * 256 + tid;        // 16-byte unit index, 8192 total
    int tileIdx = o >> 6;         // 64 units per 1KB tile
    int lt = tileIdx >> 4;        // local tile 0..7 -> (gate, sub)
    int kt = tileIdx & 15;
    int within = o & 63;
    int bl = within >> 1;         // lane within tile
    int hh = within & 1;          // which 16B half of the 32B lane payload
    int g = lt >> 1, sub = lt & 1;
    int ntg = g * 32 + slice * 2 + sub;
    const u32x4* s = (const u32x4*)(whh_p + ((size_t)(ntg * 16 + kt) * 32 + bl) * 16 + hh * 8);
    *(u32x4*)(Bs + (size_t)o * 8) = *s;
  }
  u32x4 zero4 = {0u, 0u, 0u, 0u};
  for (int i = tid; i < 2048; i += 256) ((u32x4*)As)[i] = zero4;   // h0 = 0
  for (int i = tid; i < 1024; i += 256) { cbuf[i] = 0.f; hbuf[i] = 0.f; }
  __syncthreads();

  const int g    = wave >> 1;
  const int sub  = wave & 1;
  const int nloc = sub * 16 + (lane & 15);                 // col within 32-wide slice
  const int nglobal = g * 512 + slice * 32 + nloc;         // gate-space column
  const int half = lane >> 4;
  const float biasv = bih[nglobal] + bhh[nglobal];
  const v8f zf = {0.f,0.f,0.f,0.f,0.f,0.f,0.f,0.f};

  for (int t = 0; t < TP; ++t) {
    // ---- gates(slice) = h_t @ w_hh^T(slice): 2 M-tiles x 16 K-tiles ----
    v8f acc0 = zf, acc1 = zf;
    for (int kt = 0; kt < 16; ++kt) {
      v16bf a0 = *(const v16bf*)(As + ((size_t)(kt)      * 32 + lane) * 16);
      v16bf a1 = *(const v16bf*)(As + ((size_t)(16 + kt) * 32 + lane) * 16);
      v16bf b  = *(const v16bf*)(Bs + ((size_t)(wave * 16 + kt) * 32 + lane) * 16);
      acc0 = wmma_bf16(a0, b, acc0);
      acc1 = wmma_bf16(a1, b, acc1);
    }
    // ---- add precomputed x-gates + bias, stash to LDS for recombination ----
#pragma unroll
    for (int mt = 0; mt < 2; ++mt) {
      v8f acc = mt ? acc1 : acc0;
#pragma unroll
      for (int r = 0; r < 8; ++r) {
        int b_ = mt * 16 + r + 8 * half;                   // batch index (M)
        int len = lensS[b_];
        int srow = dir ? (len - 1 - t) : t;                // reversed input indexing
        srow = srow < 0 ? 0 : srow;
        float xv = bf2f(xg[((size_t)b_ * TP + srow) * G4 + nglobal]);
        gbuf[(g * 32 + nloc) * 32 + b_] = acc[r] + xv + biasv;
      }
    }
    __syncthreads();
    // ---- cell update: each thread owns 4 (col,batch) cells ----
#pragma unroll
    for (int i = 0; i < 4; ++i) {
      int cell = tid + 256 * i;                            // 0..1023
      int col = cell >> 5, b_ = cell & 31;
      float gi = sigm(gbuf[(0 * 32 + col) * 32 + b_]);
      float gf = sigm(gbuf[(1 * 32 + col) * 32 + b_]);
      float gg = hw_tanh(gbuf[(2 * 32 + col) * 32 + b_]);
      float go = sigm(gbuf[(3 * 32 + col) * 32 + b_]);
      int len = lensS[b_];
      bool valid = t < len;
      float cn = gf * cbuf[col * 32 + b_] + gi * gg;
      float hn = go * hw_tanh(cn);
      if (valid) {
        cbuf[col * 32 + b_] = cn;
        hbuf[col * 32 + b_] = hn;
        int tout = dir ? (len - 1 - t) : t;                // un-reverse outputs
        out_cat[((size_t)b_ * TP + tout) * (2 * HD) + dir * HD + slice * 32 + col] =
            f2bf(hn);
      }
      // publish current h (old value kept when masked), double-buffered by t
      hglob[(size_t)(dir * 2 + (t & 1)) * BATCH * HD + (size_t)b_ * HD +
            slice * 32 + col] = f2bf(hbuf[col * 32 + b_]);
    }
    // ---- device-scope barrier across the 16 blocks of this direction ----
    __threadfence();
    __syncthreads();
    if (tid == 0) {
      __hip_atomic_fetch_add(&syncc[dir], 1u, __ATOMIC_RELEASE,
                             __HIP_MEMORY_SCOPE_AGENT);
      unsigned target = 16u * (unsigned)(t + 1);
      while (__hip_atomic_load(&syncc[dir], __ATOMIC_ACQUIRE,
                               __HIP_MEMORY_SCOPE_AGENT) < target)
        __builtin_amdgcn_s_sleep(2);
    }
    __syncthreads();
    __threadfence();
    // ---- gather full h_{t+1} and pack into the WMMA A layout in LDS ----
    const unsigned short* hsrc = hglob + (size_t)(dir * 2 + (t & 1)) * BATCH * HD;
#pragma unroll
    for (int i = 0; i < 4; ++i) {
      int entry = tid + 256 * i;                           // 2 mt x 16 kt x 32 lanes
      int mt = entry >> 9;
      int rem = entry & 511;
      int kt = rem >> 5;
      int L  = rem & 31;
      int m  = mt * 16 + (L & 15);
      int kb = kt * 32 + 8 * (L >> 4);
      u32x4 lo = *(const u32x4*)(hsrc + (size_t)m * HD + kb);       // K kb..kb+7
      u32x4 hi = *(const u32x4*)(hsrc + (size_t)m * HD + kb + 16);  // K kb+16..kb+23
      unsigned short* d = As + ((size_t)(mt * 16 + kt) * 32 + L) * 16;
      *(u32x4*)d = lo;
      *(u32x4*)(d + 8) = hi;
    }
    __syncthreads();
  }
}

// ---------------------------------------------------------------------------
// output = tanh(out_cat @ proj_w^T + proj_b): M=32768, K=1024, N=512, f32 out
// ---------------------------------------------------------------------------
__global__ void proj_kernel(const unsigned short* __restrict__ out_cat,
                            const unsigned short* __restrict__ proj_p,
                            const float* __restrict__ proj_b,
                            float* __restrict__ out) {
  const int lane = threadIdx.x & 31;
  const int wave = threadIdx.x >> 5;
  const int mt = blockIdx.x;
  const int half = lane >> 4;
  const int m = mt * 16 + (lane & 15);
  const unsigned short* arow = out_cat + (size_t)m * (2 * HD);

  v8f zf = {0.f,0.f,0.f,0.f,0.f,0.f,0.f,0.f};
  v8f acc[4] = {zf, zf, zf, zf};
  const int ntb = wave * 4;

  for (int kt = 0; kt < 32; ++kt) {
    int kb = kt * 32 + 8 * half;
    v8us lo = *(const v8us*)(arow + kb);
    v8us hi = *(const v8us*)(arow + kb + 16);
    v16us ua;
#pragma unroll
    for (int e = 0; e < 8; ++e) { ua[e] = lo[e]; ua[8 + e] = hi[e]; }
    v16bf a = __builtin_bit_cast(v16bf, ua);
#pragma unroll
    for (int j = 0; j < 4; ++j) {
      v16bf b = *(const v16bf*)(proj_p + ((size_t)((ntb + j) * 32 + kt) * 32 + lane) * 16);
      acc[j] = wmma_bf16(a, b, acc[j]);
    }
  }
#pragma unroll
  for (int j = 0; j < 4; ++j) {
    int n = (ntb + j) * 16 + (lane & 15);
    float bb = proj_b[n];
#pragma unroll
    for (int r = 0; r < 8; ++r) {
      int mr = mt * 16 + r + 8 * half;
      out[(size_t)mr * PD + n] = hw_tanh(acc[j][r] + bb);
    }
  }
}

// ---------------------------------------------------------------------------
extern "C" void kernel_launch(void* const* d_in, const int* in_sizes, int n_in,
                              void* d_out, int out_size, void* d_ws, size_t ws_size,
                              hipStream_t stream) {
  (void)in_sizes; (void)n_in; (void)out_size; (void)ws_size;
  const float* input_data = (const float*)d_in[0];
  const int*   input_size = (const int*)d_in[1];
  const float* w_ih_f = (const float*)d_in[2];
  const float* w_hh_f = (const float*)d_in[3];
  const float* b_ih_f = (const float*)d_in[4];
  const float* b_hh_f = (const float*)d_in[5];
  const float* w_ih_r = (const float*)d_in[6];
  const float* w_hh_r = (const float*)d_in[7];
  const float* b_ih_r = (const float*)d_in[8];
  const float* b_hh_r = (const float*)d_in[9];
  const float* proj_w = (const float*)d_in[10];
  const float* proj_b = (const float*)d_in[11];

  // workspace layout (bytes); total ~349 MB
  char* ws = (char*)d_ws;
  size_t off = 0;
  unsigned short* xg     = (unsigned short*)(ws + off); off += (size_t)2 * MROWS * G4 * 2;
  unsigned short* outcat = (unsigned short*)(ws + off); off += (size_t)MROWS * 2 * HD * 2;
  unsigned short* whh_p  = (unsigned short*)(ws + off); off += (size_t)2 * 128 * 16 * 512 * 2;
  unsigned short* wih_p  = (unsigned short*)(ws + off); off += (size_t)2 * 128 * 32 * 512 * 2;
  unsigned short* proj_p = (unsigned short*)(ws + off); off += (size_t)32 * 32 * 512 * 2;
  unsigned short* hglob  = (unsigned short*)(ws + off); off += (size_t)4 * BATCH * HD * 2;
  unsigned int*   syncc  = (unsigned int*)(ws + off);   off += 256;

  float* out = (float*)d_out;
  float* lens_out = out + (size_t)MROWS * PD;

  // 1) zero out_cat, reset barrier counters, emit lens
  init_kernel<<<2048, 256, 0, stream>>>((unsigned int*)outcat,
                                        (size_t)MROWS * 2 * HD * 2 / 4,
                                        syncc, input_size, lens_out);
  // 2) pack weights into WMMA B layouts (bf16)
  pack_b_kernel<<<(128 * 16 * 32) / 256, 256, 0, stream>>>(w_hh_f, whh_p, 128, 16, 512);
  pack_b_kernel<<<(128 * 16 * 32) / 256, 256, 0, stream>>>(w_hh_r, whh_p + 128 * 16 * 512, 128, 16, 512);
  pack_b_kernel<<<(128 * 32 * 32) / 256, 256, 0, stream>>>(w_ih_f, wih_p, 128, 32, 1024);
  pack_b_kernel<<<(128 * 32 * 32) / 256, 256, 0, stream>>>(w_ih_r, wih_p + 128 * 32 * 512, 128, 32, 1024);
  pack_b_kernel<<<(32 * 32 * 32) / 256, 256, 0, stream>>>(proj_w, proj_p, 32, 32, 1024);
  // 3) big input GEMMs: xg_d = x @ w_ih_d^T   (x = contiguous view of input)
  xgemm_kernel<<<8192, 256, 0, stream>>>(input_data, wih_p, xg);
  xgemm_kernel<<<8192, 256, 0, stream>>>(input_data, wih_p + 128 * 32 * 512,
                                         xg + (size_t)MROWS * G4);
  // 4) persistent recurrence (both directions), 188KB LDS per block
  hipFuncSetAttribute((const void*)lstm_rec_kernel,
                      hipFuncAttributeMaxDynamicSharedMemorySize, REC_SMEM);
  lstm_rec_kernel<<<32, 256, REC_SMEM, stream>>>(xg, whh_p, b_ih_f, b_hh_f,
                                                 b_ih_r, b_hh_r, input_size,
                                                 outcat, hglob, syncc);
  // 5) fused projection + tanh
  proj_kernel<<<2048, 256, 0, stream>>>(outcat, proj_p, proj_b, out);
}
